// SS2D_16716012716799
// MI455X (gfx1250) — matrix operand order repaired and checked by reading, not compiled, not run on hardware
//
#include <hip/hip_runtime.h>
#include <hip/hip_bf16.h>

// ---------------------------------------------------------------------------
// SS2D (VMamba) block for MI455X / gfx1250, wave32, WMMA bf16 + async-LDS path.
// ---------------------------------------------------------------------------

#define BATCH   8
#define HH      64
#define WW      64
#define CMODEL  96
#define LSEQ    4096            // HH*WW
#define DINNER  192
#define NSTATE  16
#define DTRANK  6
#define KDIR    4
#define LN_EPS  1e-5f
#define LOG2E   1.4426950408889634f

typedef __attribute__((ext_vector_type(16))) __bf16 v16bf;
typedef __attribute__((ext_vector_type(8)))  __bf16 v8bf;
typedef __attribute__((ext_vector_type(4)))  __bf16 v4bf;
typedef __attribute__((ext_vector_type(8)))  float  v8f;

// workspace byte offsets (16B aligned)
#define OFF_WIN    0u                    // 384*96   bf16  (in_proj_w)
#define OFF_WOUT   73728u                // 96*192   bf16  (out_proj_w)
#define OFF_A2     110592u               // 192*16   f32   (-exp(A_log)*log2e)
#define OFF_XD     122880u               // B*192*L  f32   x_  (channel-major)
#define OFF_Z      25288704u             // B*L*192  f32   z   (pixel-major)
#define OFF_XC     50454528u             // B*192*L  f32   conv+SiLU output
#define OFF_XDBL   75620352u             // B*K*38*L f32   x_dbl (dts|Bs|Cs)
#define OFF_Y      95543296u             // B*L*192  f32   merged scan output

#define NDBL (DTRANK + 2*NSTATE)         // 38

// ------------------- CDNA5 async global->LDS helpers -----------------------
#if defined(__gfx1250__) && \
    __has_builtin(__builtin_amdgcn_global_load_async_to_lds_b32) && \
    __has_builtin(__builtin_amdgcn_global_load_async_to_lds_b128)
#define HAVE_ASYNC_LDS 1
#else
#define HAVE_ASYNC_LDS 0
#endif

#if HAVE_ASYNC_LDS
typedef int i32x4 __attribute__((vector_size(16)));
typedef __attribute__((address_space(1))) int   g_i32;
typedef __attribute__((address_space(3))) int   l_i32;
typedef __attribute__((address_space(1))) i32x4 g_i32x4;
typedef __attribute__((address_space(3))) i32x4 l_i32x4;

__device__ __forceinline__ void async_g2l_b32(const void* g, const void* l) {
  __builtin_amdgcn_global_load_async_to_lds_b32(
      (g_i32*)(unsigned long long)g,
      (l_i32*)(unsigned int)(unsigned long long)l, 0, 0);
}
__device__ __forceinline__ void async_g2l_b128(const void* g, const void* l) {
  __builtin_amdgcn_global_load_async_to_lds_b128(
      (g_i32x4*)(unsigned long long)g,
      (l_i32x4*)(unsigned int)(unsigned long long)l, 0, 0);
}
__device__ __forceinline__ void wait_async0() {
#if __has_builtin(__builtin_amdgcn_s_wait_asynccnt)
  __builtin_amdgcn_s_wait_asynccnt(0);
#else
  asm volatile("s_wait_asynccnt 0x0" ::: "memory");
#endif
}
#endif

__device__ __forceinline__ int dirmap(int k, int l) {
  int t = (k & 2) ? (LSEQ - 1 - l) : l;
  if (k & 1) t = ((t & 63) << 6) | (t >> 6);   // (h,w) <-> (w,h) transpose, H=W=64
  return t;
}

__device__ __forceinline__ v16bf cat16(v8bf lo, v8bf hi) {
  return __builtin_shufflevector(lo, hi, 0,1,2,3,4,5,6,7,8,9,10,11,12,13,14,15);
}

// ---------------------------------------------------------------------------
// K0: pack weights to bf16, precompute A2 = -exp(A_log)*log2e
// ---------------------------------------------------------------------------
__global__ __launch_bounds__(256) void k0_prep(
    const float* __restrict__ in_proj_w, const float* __restrict__ out_proj_w,
    const float* __restrict__ A_log, char* __restrict__ ws) {
  __bf16* win  = (__bf16*)(ws + OFF_WIN);
  __bf16* wout = (__bf16*)(ws + OFF_WOUT);
  float*  a2   = (float*)(ws + OFF_A2);
  int g = blockIdx.x * 256 + threadIdx.x;
  if (g < 384*96)                       win[g] = (__bf16)in_proj_w[g];
  else if (g < 384*96 + 96*192)         wout[g - 384*96] = (__bf16)out_proj_w[g - 384*96];
  else if (g < 384*96 + 96*192 + DINNER*NSTATE) {
    int i = g - (384*96 + 96*192);
    a2[i] = -__expf(A_log[i]) * LOG2E;
  }
}

// ---------------------------------------------------------------------------
// K1: in_proj GEMM  xz[m][n] = x[m][:96] . Win[n][:96]   (WMMA bf16)
//     n<192  -> x_  stored channel-major (B, D, L)
//     n>=192 -> z   stored pixel-major  (B, L, D)
// grid (512, 12), block 256 (8 waves). block tile: 64 rows x 32 cols.
// ---------------------------------------------------------------------------
__global__ __launch_bounds__(256) void k1_inproj(
    const float* __restrict__ x, char* __restrict__ ws) {
  __shared__ __align__(16) __bf16 xt[64 * 96];
  const __bf16* win = (const __bf16*)(ws + OFF_WIN);
  float* xd = (float*)(ws + OFF_XD);
  float* zb = (float*)(ws + OFF_Z);

  const int tid = threadIdx.x;
  const int m0  = blockIdx.x * 64;

  // stage 64x96 f32 tile of x -> bf16 LDS
  for (int f4 = tid; f4 < (64*96)/4; f4 += 256) {
    const float4 v = *(const float4*)(x + m0*96 + f4*4);
    v4bf p; p[0]=(__bf16)v.x; p[1]=(__bf16)v.y; p[2]=(__bf16)v.z; p[3]=(__bf16)v.w;
    *(v4bf*)(xt + f4*4) = p;
  }
  __syncthreads();

  const int wid  = tid >> 5, lane = tid & 31;
  const int mi   = wid & 3,  ni   = wid >> 2;
  const int khalf = lane >> 4, l15 = lane & 15;
  const int ncol  = blockIdx.y * 32 + ni * 16 + l15;

  v8f acc = {};
  #pragma unroll
  for (int kc = 0; kc < 96; kc += 32) {
    const int ao = (mi*16 + l15) * 96 + kc + khalf * 8;
    v16bf a = cat16(*(const v8bf*)(xt + ao), *(const v8bf*)(xt + ao + 16));
    const __bf16* bp = win + ncol * 96 + kc + khalf * 8;
    v16bf b = cat16(*(const v8bf*)bp, *(const v8bf*)(bp + 16));
    acc = __builtin_amdgcn_wmma_f32_16x16x32_bf16(false, a, false, b,
                                                  (short)0, acc, false, false);
  }

  const int mbase = m0 + mi*16 + khalf*8;       // 8 consecutive rows (same batch)
  const int b = mbase >> 12, l0 = mbase & 4095;
  if (ncol < DINNER) {                          // x_ : (B, D, L) contiguous in l
    float* p = xd + ((size_t)(b*DINNER + ncol)) * LSEQ + l0;
    *(float4*)(p)     = make_float4(acc[0], acc[1], acc[2], acc[3]);
    *(float4*)(p + 4) = make_float4(acc[4], acc[5], acc[6], acc[7]);
  } else {                                      // z : (B, L, D)
    const int n = ncol - DINNER;
    #pragma unroll
    for (int i = 0; i < 8; ++i)
      zb[((size_t)(b*LSEQ + l0 + i)) * DINNER + n] = acc[i];
  }
}

// ---------------------------------------------------------------------------
// K2: depthwise 3x3 conv + bias + SiLU on x_ (B,192,64,64) -> xc
// ---------------------------------------------------------------------------
__global__ __launch_bounds__(256) void k2_conv(
    const float* __restrict__ conv_w, const float* __restrict__ conv_b,
    char* __restrict__ ws) {
  const float* xd = (const float*)(ws + OFF_XD);
  float* xc = (float*)(ws + OFF_XC);
  const int g = blockIdx.x * 256 + threadIdx.x;
  const int w = g & 63, h = (g >> 6) & 63, plane = g >> 12;   // plane uniform per block
  const int d = plane % DINNER;
  const float* src = xd + (size_t)plane * LSEQ;
  const float* wt  = conv_w + d * 9;
  float acc = conv_b[d];
  #pragma unroll
  for (int dy = -1; dy <= 1; ++dy) {
    const int hy = h + dy;
    if ((unsigned)hy < 64u) {
      #pragma unroll
      for (int dx = -1; dx <= 1; ++dx) {
        const int wx = w + dx;
        if ((unsigned)wx < 64u)
          acc = fmaf(wt[(dy+1)*3 + (dx+1)], src[hy*64 + wx], acc);
      }
    }
  }
  xc[(size_t)plane * LSEQ + h*64 + w] = acc / (1.f + __expf(-acc));   // SiLU
}

// ---------------------------------------------------------------------------
// K3: x_dbl[b,k,c,l] = sum_d Wx[k,c,d] * xc[b,d,dirmap(k,l)]
// grid (L/128, 38, B*K), block 128; coalesced over l, W broadcast via L2.
// ---------------------------------------------------------------------------
__global__ __launch_bounds__(128) void k3_xdbl(
    const float* __restrict__ x_proj_w, char* __restrict__ ws) {
  const float* xc = (const float*)(ws + OFF_XC);
  float* xdbl = (float*)(ws + OFF_XDBL);
  const int bk = blockIdx.z, b = bk >> 2, k = bk & 3;
  const int c = blockIdx.y;
  const int l = blockIdx.x * 128 + threadIdx.x;
  const int m = dirmap(k, l);
  const float* wrow = x_proj_w + ((size_t)(k*NDBL + c)) * DINNER;
  const float* xb = xc + (size_t)b * DINNER * LSEQ + m;
  float acc = 0.f;
  #pragma unroll 8
  for (int d = 0; d < DINNER; ++d)
    acc = fmaf(wrow[d], xb[(size_t)d * LSEQ], acc);
  xdbl[(((size_t)bk * NDBL) + c) * LSEQ + l] = acc;
}

// ---------------------------------------------------------------------------
// K4: zero the merge buffer (atomic-accumulated each call)
// ---------------------------------------------------------------------------
__global__ __launch_bounds__(256) void k4_zero(char* __restrict__ ws) {
  float4* y = (float4*)(ws + OFF_Y);
  y[blockIdx.x * 256 + threadIdx.x] = make_float4(0.f, 0.f, 0.f, 0.f);
}

// ---------------------------------------------------------------------------
// K5: selective scan. One block per (b,k); lane = channel d; h[16] in VGPRs.
// Chunks of 64 timesteps staged into LDS with GLOBAL_LOAD_ASYNC_TO_LDS
// (ASYNCcnt path); u rows padded to stride 65 -> conflict-free broadcasts.
// Cross-merge fused via global f32 atomics into y (B,L,192).
// ---------------------------------------------------------------------------
__global__ __launch_bounds__(192) void k5_scan(
    const float* __restrict__ dt_projs_w, const float* __restrict__ dt_projs_b,
    const float* __restrict__ Ds, char* __restrict__ ws) {
  __shared__ float uld[DINNER * 65];
  __shared__ __align__(16) float dbl[NDBL * 64];
  const float* xc   = (const float*)(ws + OFF_XC);
  const float* xdbl = (const float*)(ws + OFF_XDBL);
  const float* a2   = (const float*)(ws + OFF_A2);
  float* yout = (float*)(ws + OFF_Y);

  const int bk = blockIdx.x, b = bk >> 2, k = bk & 3;
  const int d = threadIdx.x;

  float A2[NSTATE], h[NSTATE], dtw[DTRANK];
  #pragma unroll
  for (int n = 0; n < NSTATE; ++n) { A2[n] = a2[d*NSTATE + n]; h[n] = 0.f; }
  #pragma unroll
  for (int r = 0; r < DTRANK; ++r) dtw[r] = dt_projs_w[(k*DINNER + d)*DTRANK + r];
  const float dtb   = dt_projs_b[k*DINNER + d];
  const float dskip = Ds[k*DINNER + d];

  const float* xcrow = xc + ((size_t)(b*DINNER + d)) * LSEQ;
  const float* dblb  = xdbl + (size_t)bk * NDBL * LSEQ;
  float* yb = yout + (size_t)b * LSEQ * DINNER + d;

  for (int t0 = 0; t0 < LSEQ; t0 += 64) {
#if HAVE_ASYNC_LDS
    // stage dts|B|C chunk (38 rows x 64): one b128 per float4, straight to LDS
    for (int f4 = d; f4 < NDBL*16; f4 += DINNER)
      async_g2l_b128(dblb + (size_t)(f4 >> 4) * LSEQ + t0 + (f4 & 15) * 4,
                     &dbl[(f4 >> 4) * 64 + (f4 & 15) * 4]);
    // stage this lane's u row (direction-dependent gather), b32 to LDS
    #pragma unroll 4
    for (int j = 0; j < 64; ++j)
      async_g2l_b32(xcrow + dirmap(k, t0 + j), &uld[d*65 + j]);
    if (t0 + 64 < LSEQ)
      __builtin_prefetch(dblb + t0 + 64, 0, 3);
    wait_async0();
#else
    for (int idx = d; idx < NDBL*64; idx += DINNER)
      dbl[idx] = dblb[(size_t)(idx >> 6) * LSEQ + t0 + (idx & 63)];
    #pragma unroll 4
    for (int j = 0; j < 64; ++j)
      uld[d*65 + j] = xcrow[dirmap(k, t0 + j)];
    if (t0 + 64 < LSEQ)
      __builtin_prefetch(dblb + t0 + 64, 0, 3);
#endif
    __syncthreads();

    for (int j = 0; j < 64; ++j) {
      const float u = uld[d*65 + j];
      float xv = dtb;
      #pragma unroll
      for (int r = 0; r < DTRANK; ++r) xv = fmaf(dtw[r], dbl[r*64 + j], xv);
      const float delta = (xv > 20.f) ? xv : __logf(1.f + __expf(xv));
      const float du = delta * u;
      float y = dskip * u;
      #pragma unroll
      for (int n = 0; n < NSTATE; ++n) {
        const float e = __builtin_amdgcn_exp2f(delta * A2[n]);
        h[n] = fmaf(e, h[n], du * dbl[(DTRANK + n)*64 + j]);
        y = fmaf(h[n], dbl[(DTRANK + NSTATE + n)*64 + j], y);
      }
      atomicAdd(yb + (size_t)dirmap(k, t0 + j) * DINNER, y);   // cross-merge
    }
    __syncthreads();
  }
}

// ---------------------------------------------------------------------------
// K6: LayerNorm(192) + SiLU-gate + out_proj (WMMA bf16, 64 pixels/block)
// ---------------------------------------------------------------------------
__global__ __launch_bounds__(256) void k6_final(
    const float* __restrict__ lnw, const float* __restrict__ lnb,
    char* __restrict__ ws, float* __restrict__ out) {
  __shared__ __align__(16) __bf16 gt[64 * DINNER];
  __shared__ float rs1[256], rs2[256], mus[64], rsd[64];
  const float* yb = (const float*)(ws + OFF_Y);
  const float* zb = (const float*)(ws + OFF_Z);
  const __bf16* wout = (const __bf16*)(ws + OFF_WOUT);

  const int tid = threadIdx.x;
  const int pl = tid >> 2, q = tid & 3;
  const int pixel = blockIdx.x * 64 + pl;

  // --- LN stats: 4 threads x 48 channels per pixel ---
  float yv[48], s = 0.f, s2 = 0.f;
  const float* yr = yb + (size_t)pixel * DINNER + q * 48;
  #pragma unroll 8
  for (int i = 0; i < 48; ++i) { yv[i] = yr[i]; s += yv[i]; s2 = fmaf(yv[i], yv[i], s2); }
  rs1[tid] = s; rs2[tid] = s2;
  __syncthreads();
  if (q == 0) {
    const int t4 = pl * 4;
    const float sa = rs1[t4] + rs1[t4+1] + rs1[t4+2] + rs1[t4+3];
    const float sb = rs2[t4] + rs2[t4+1] + rs2[t4+2] + rs2[t4+3];
    const float mu = sa * (1.f / DINNER);
    mus[pl] = mu;
    rsd[pl] = rsqrtf(sb * (1.f / DINNER) - mu * mu + LN_EPS);
  }
  __syncthreads();

  // --- normalize + gate -> bf16 A-tile ---
  {
    const float mu = mus[pl], r = rsd[pl];
    const float* zr = zb + (size_t)pixel * DINNER + q * 48;
    #pragma unroll 8
    for (int i = 0; i < 48; ++i) {
      const int c = q * 48 + i;
      const float z = zr[i];
      float g = fmaf((yv[i] - mu) * r, lnw[c], lnb[c]);
      g *= z / (1.f + __expf(-z));
      gt[pl * DINNER + c] = (__bf16)g;
    }
  }
  __syncthreads();

  // --- out_proj: 4 M-tiles x 6 N-tiles, 8 waves x 3 tiles each ---
  const int wid = tid >> 5, lane = tid & 31;
  const int khalf = lane >> 4, l15 = lane & 15;
  #pragma unroll
  for (int tt = 0; tt < 3; ++tt) {
    const int tile = wid * 3 + tt, mt = tile / 6, nt = tile % 6;
    const int ncol = nt * 16 + l15;
    v8f acc = {};
    #pragma unroll
    for (int kc = 0; kc < DINNER; kc += 32) {
      const int ao = (mt*16 + l15) * DINNER + kc + khalf * 8;
      v16bf a = cat16(*(const v8bf*)(gt + ao), *(const v8bf*)(gt + ao + 16));
      const __bf16* bp = wout + ncol * DINNER + kc + khalf * 8;
      v16bf bmat = cat16(*(const v8bf*)bp, *(const v8bf*)(bp + 16));
      acc = __builtin_amdgcn_wmma_f32_16x16x32_bf16(false, a, false, bmat,
                                                    (short)0, acc, false, false);
    }
    const int mbase = blockIdx.x * 64 + mt*16 + khalf*8;
    #pragma unroll
    for (int i = 0; i < 8; ++i)
      out[(size_t)(mbase + i) * CMODEL + ncol] = acc[i];
  }
}

// ---------------------------------------------------------------------------
extern "C" void kernel_launch(void* const* d_in, const int* in_sizes, int n_in,
                              void* d_out, int out_size, void* d_ws, size_t ws_size,
                              hipStream_t stream) {
  const float* x          = (const float*)d_in[0];
  const float* in_proj_w  = (const float*)d_in[1];
  const float* conv_w     = (const float*)d_in[2];
  const float* conv_b     = (const float*)d_in[3];
  const float* x_proj_w   = (const float*)d_in[4];
  const float* dt_projs_w = (const float*)d_in[5];
  const float* dt_projs_b = (const float*)d_in[6];
  const float* A_log      = (const float*)d_in[7];
  const float* Ds         = (const float*)d_in[8];
  const float* out_norm_w = (const float*)d_in[9];
  const float* out_norm_b = (const float*)d_in[10];
  const float* out_proj_w = (const float*)d_in[11];
  float* out = (float*)d_out;
  char* ws = (char*)d_ws;

  k0_prep<<<(384*96 + 96*192 + DINNER*NSTATE + 255)/256, 256, 0, stream>>>(
      in_proj_w, out_proj_w, A_log, ws);
  k1_inproj<<<dim3((BATCH*LSEQ)/64, (2*DINNER)/32), 256, 0, stream>>>(x, ws);
  k2_conv<<<(BATCH*DINNER*LSEQ)/256, 256, 0, stream>>>(conv_w, conv_b, ws);
  k3_xdbl<<<dim3(LSEQ/128, NDBL, BATCH*KDIR), 128, 0, stream>>>(x_proj_w, ws);
  k4_zero<<<(BATCH*LSEQ*DINNER)/(256*4), 256, 0, stream>>>(ws);
  k5_scan<<<BATCH*KDIR, DINNER, 0, stream>>>(dt_projs_w, dt_projs_b, Ds, ws);
  k6_final<<<(BATCH*LSEQ)/64, 256, 0, stream>>>(out_norm_w, out_norm_b, ws, out);
  (void)in_sizes; (void)n_in; (void)out_size; (void)ws_size;
}
